// AttentionBlock_38577396253211
// MI455X (gfx1250) — matrix operand order repaired and checked by reading, not compiled, not run on hardware
//
#include <hip/hip_runtime.h>

typedef __attribute__((ext_vector_type(16))) _Float16 v16h;
typedef __attribute__((ext_vector_type(8)))  float    v8f;

#define SEQ   1024
#define CDIM  512
#define DK    128
#define NH    4
#define BATCH 16
#define NQKV  1536   // 3 * NH * DK

__device__ __forceinline__ v8f wmma_f32_f16(v16h a, v16h b, v8f c) {
  // 8 args: (neg_a, A, neg_b, B, c_mod, C, reuse_a, reuse_b)
  return __builtin_amdgcn_wmma_f32_16x16x32_f16(false, a, false, b, (short)0, c,
                                                false, false);
}

// wave index as a proven-uniform scalar (SGPR) -> tile/routing math on SALU
__device__ __forceinline__ int wave_id() {
  return __builtin_amdgcn_readfirstlane((int)(threadIdx.x >> 5));
}

// ---------------- fragment load helpers (wave32, contiguous k = 16*(lane/16) + t) -----------
// A 16x32 f16 fragment: A[m][k] = base[m*lda + k]  (row-major source)
__device__ __forceinline__ v16h load_a_frag(const _Float16* base, int lda, int lane) {
  v16h a;
  const int m  = lane & 15;
  const int kb = (lane < 16) ? 0 : 16;
  const _Float16* p = base + (size_t)m * lda;
#pragma unroll
  for (int t = 0; t < 16; ++t) {
    a[t]     = p[kb + t];
  }
  return a;
}

// A 16x32 fragment from fp32 source accessed TRANSPOSED: A[m][k] = src[k*lds + m]
__device__ __forceinline__ v16h load_aT_frag_f32(const float* src, int lds, int lane) {
  v16h a;
  const int m  = lane & 15;
  const int kb = (lane < 16) ? 0 : 16;
  const float* p = src + m;
#pragma unroll
  for (int t = 0; t < 16; ++t) {
    a[t]     = (_Float16)p[(size_t)(kb + t) * lds];
  }
  return a;
}

// B 32x16 fragment, transposed source: B[k][n] = src[n*ldn + k]
__device__ __forceinline__ v16h load_bT_frag(const _Float16* src, int ldn, int lane) {
  v16h b;
  const int n  = lane & 15;
  const int kb = (lane < 16) ? 0 : 16;
  const _Float16* p = src + (size_t)n * ldn + kb;
#pragma unroll
  for (int t = 0; t < 16; ++t) b[t] = p[t];
  return b;
}

// B 32x16 fragment, direct source: B[k][n] = src[k*ldb + n]
__device__ __forceinline__ v16h load_b_frag(const _Float16* src, int ldb, int lane) {
  v16h b;
  const int n  = lane & 15;
  const int kb = (lane < 16) ? 0 : 16;
#pragma unroll
  for (int t = 0; t < 16; ++t) b[t] = src[(size_t)(kb + t) * ldb + n];
  return b;
}

// ---------------- pass 1: fp32 -> f16 weight conversion --------------------
__global__ void cvt_weights(const float* __restrict__ wp, const float* __restrict__ wo,
                            _Float16* __restrict__ wph, _Float16* __restrict__ woh) {
  int i = blockIdx.x * blockDim.x + threadIdx.x;
  if (i < NQKV * CDIM) wph[i] = (_Float16)wp[i];
  if (i < CDIM * CDIM) woh[i] = (_Float16)wo[i];
}

// ---------------- pass 2: QKV projection (WMMA GEMM, bias fused) -----------
// Each wave owns a 16x64 tile: one A fragment feeds 4 WMMAs.
// qkv[m][n] = sum_k xs[m][k]*w_proj[n][k] + b_proj[n];  xs[m][k] = x[b][k][s]
__global__ void qkv_proj(const float* __restrict__ x,
                         const _Float16* __restrict__ wph,
                         const float* __restrict__ bproj,
                         _Float16* __restrict__ qh,
                         _Float16* __restrict__ kh,
                         _Float16* __restrict__ vh) {
  const int lane = threadIdx.x & 31;
  const int wave = wave_id();                       // SGPR-uniform
  const int m0 = blockIdx.x * 16;
  const int n0 = (blockIdx.y * 4 + wave) * 64;      // scalar
  const int b  = m0 / SEQ;
  const int s0 = m0 % SEQ;

  const float* xbase = x + (size_t)b * CDIM * SEQ + s0;
  v8f acc[4] = {};
  for (int k0 = 0; k0 < CDIM; k0 += 32) {
    v16h a = load_aT_frag_f32(xbase + (size_t)k0 * SEQ, SEQ, lane);
#pragma unroll
    for (int t = 0; t < 4; ++t) {
      v16h bm = load_bT_frag(wph + (size_t)(n0 + t * 16) * CDIM + k0, CDIM, lane);
      acc[t] = wmma_f32_f16(a, bm, acc[t]);
    }
  }

  const int rbase = (lane < 16) ? 0 : 8;
#pragma unroll
  for (int t = 0; t < 4; ++t) {
    // all routing operands are scalar -> SALU magic-divides, once per wave
    const int ns    = n0 + t * 16;
    const int h     = ns / (3 * DK);
    const int r     = ns % (3 * DK);
    const int which = r / DK;
    const int d0    = r % DK;
    _Float16* dst = (which == 0) ? qh : (which == 1) ? kh : vh;
    dst += ((size_t)(b * NH + h) * SEQ) * DK + d0 + (lane & 15);
    const float bias = bproj[ns + (lane & 15)];
#pragma unroll
    for (int rr = 0; rr < 8; ++rr) {
      int s = s0 + rbase + rr;
      dst[(size_t)s * DK] = (_Float16)(acc[t][rr] + bias);
    }
  }
}

// ---------------- pass 3: column-wise softmax stats ------------------------
// Reference softmax is over the QUERY axis i: stats are per (b,h,j).
// K fragments for the wave's 16 columns are loop-invariant -> hoisted.
__global__ void attn_stats(const _Float16* __restrict__ qh,
                           const _Float16* __restrict__ kh,
                           float* __restrict__ colmax,
                           float* __restrict__ colsum) {
  const int lane = threadIdx.x & 31;
  const int wave = wave_id();
  const int bh = blockIdx.x;
  const int j0 = (blockIdx.y * 4 + wave) * 16;      // scalar
  const _Float16* qb = qh + (size_t)bh * SEQ * DK;
  const _Float16* kb = kh + (size_t)bh * SEQ * DK;
  const float scale = 0.08838834764831845f;   // 1/sqrt(128)

  v16h kf[4];
#pragma unroll
  for (int kk = 0; kk < 4; ++kk)
    kf[kk] = load_bT_frag(kb + (size_t)j0 * DK + kk * 32, DK, lane);

  float mx = -3.4e38f;
  for (int i0 = 0; i0 < SEQ; i0 += 16) {
    v8f s = {};
#pragma unroll
    for (int kk = 0; kk < 4; ++kk) {
      v16h a = load_a_frag(qb + (size_t)i0 * DK + kk * 32, DK, lane);
      s = wmma_f32_f16(a, kf[kk], s);
    }
#pragma unroll
    for (int r = 0; r < 8; ++r) mx = fmaxf(mx, s[r] * scale);
  }
  mx = fmaxf(mx, __shfl_xor(mx, 16, 32));   // combine row-halves (same column)

  float sum = 0.f;
  for (int i0 = 0; i0 < SEQ; i0 += 16) {
    v8f s = {};
#pragma unroll
    for (int kk = 0; kk < 4; ++kk) {
      v16h a = load_a_frag(qb + (size_t)i0 * DK + kk * 32, DK, lane);
      s = wmma_f32_f16(a, kf[kk], s);
    }
#pragma unroll
    for (int r = 0; r < 8; ++r) sum += __expf(s[r] * scale - mx);
  }
  sum += __shfl_xor(sum, 16, 32);

  if (lane < 16) {
    colmax[(size_t)bh * SEQ + j0 + lane] = mx;
    colsum[(size_t)bh * SEQ + j0 + lane] = sum;
  }
}

// ---------------- pass 4: O = softmax_cols(S) @ V --------------------------
// One wave owns a full 16x128 O strip (8 accumulators): S for the i-tile is
// computed ONCE and the LDS-repacked P fragment feeds all 8 V column tiles.
__global__ void attn_pv(const _Float16* __restrict__ qh,
                        const _Float16* __restrict__ kh,
                        const _Float16* __restrict__ vh,
                        const float* __restrict__ colmax,
                        const float* __restrict__ colsum,
                        _Float16* __restrict__ oh) {
  __shared__ _Float16 stile[4][16][32];       // per-wave P staging (C->A repack)
  const int lane = threadIdx.x & 31;
  const int wave = wave_id();
  const int bh = blockIdx.y;
  const int i0 = (blockIdx.x * 4 + wave) * 16;      // scalar
  const int b  = bh / NH;
  const int h  = bh % NH;
  const _Float16* qb = qh + (size_t)bh * SEQ * DK;
  const _Float16* kb = kh + (size_t)bh * SEQ * DK;
  const _Float16* vb = vh + (size_t)bh * SEQ * DK;
  const float* cmb = colmax + (size_t)bh * SEQ;
  const float* csb = colsum + (size_t)bh * SEQ;
  const float scale = 0.08838834764831845f;

  // Q fragments for this i-tile are invariant over the whole j loop
  v16h qf[4];
#pragma unroll
  for (int kk = 0; kk < 4; ++kk)
    qf[kk] = load_a_frag(qb + (size_t)i0 * DK + kk * 32, DK, lane);

  v8f acc[8] = {};
  for (int j0 = 0; j0 < SEQ; j0 += 32) {
#pragma unroll
    for (int half = 0; half < 2; ++half) {     // two 16x16 score tiles
      const int jc = j0 + half * 16;
      v8f s = {};
#pragma unroll
      for (int kk = 0; kk < 4; ++kk) {
        v16h bm = load_bT_frag(kb + (size_t)jc * DK + kk * 32, DK, lane);
        s = wmma_f32_f16(qf[kk], bm, s);
      }
      const int   c  = lane & 15;
      const float cm = cmb[jc + c];
      const float rs = 1.f / csb[jc + c];
      const int rbase = (lane < 16) ? 0 : 8;
#pragma unroll
      for (int r = 0; r < 8; ++r) {
        float p = __expf(s[r] * scale - cm) * rs;
        stile[wave][rbase + r][half * 16 + c] = (_Float16)p;
      }
    }
    __syncthreads();
    v16h a = load_a_frag(&stile[wave][0][0], 32, lane);
#pragma unroll
    for (int dt = 0; dt < 8; ++dt) {           // one P fragment feeds 8 WMMAs
      v16h vm = load_b_frag(vb + (size_t)j0 * DK + dt * 16, DK, lane);
      acc[dt] = wmma_f32_f16(a, vm, acc[dt]);
    }
    __syncthreads();
  }

  // store O (f16) as [b, s, h*DK + d] for the final GEMM
  const int rbase = (lane < 16) ? 0 : 8;
#pragma unroll
  for (int dt = 0; dt < 8; ++dt) {
    const int d = dt * 16 + (lane & 15);
    _Float16* dst = oh + (size_t)b * SEQ * (NH * DK) + (size_t)h * DK + d;
#pragma unroll
    for (int r = 0; r < 8; ++r) {
      int s = i0 + rbase + r;
      dst[(size_t)s * (NH * DK)] = (_Float16)acc[dt][r];
    }
  }
}

// ---------------- pass 5: out projection + bias + residual + transpose -----
// Each wave owns a 16x64 tile: one A fragment feeds 4 WMMAs.
__global__ void out_proj(const _Float16* __restrict__ oh,
                         const _Float16* __restrict__ woh,
                         const float* __restrict__ bout,
                         const float* __restrict__ x,
                         float* __restrict__ out) {
  const int lane = threadIdx.x & 31;
  const int wave = wave_id();
  const int m0 = blockIdx.x * 16;
  const int n0 = (blockIdx.y * 4 + wave) * 64;      // scalar
  const int b  = m0 / SEQ;
  const int s0 = m0 % SEQ;

  v8f acc[4] = {};
  for (int k0 = 0; k0 < CDIM; k0 += 32) {
    v16h a = load_a_frag(oh + (size_t)m0 * CDIM + k0, CDIM, lane);
#pragma unroll
    for (int t = 0; t < 4; ++t) {
      v16h bm = load_bT_frag(woh + (size_t)(n0 + t * 16) * CDIM + k0, CDIM, lane);
      acc[t] = wmma_f32_f16(a, bm, acc[t]);
    }
  }

  const int rbase = (lane < 16) ? 0 : 8;
#pragma unroll
  for (int t = 0; t < 4; ++t) {
    const int   c    = n0 + t * 16 + (lane & 15);
    const float bias = bout[c];
    const float* xcol = x   + ((size_t)b * CDIM + c) * SEQ + s0;
    float*       ocol = out + ((size_t)b * CDIM + c) * SEQ + s0;
#pragma unroll
    for (int r = 0; r < 8; ++r) {
      int s = rbase + r;
      ocol[s] = acc[t][r] + bias + xcol[s];   // residual xs + transpose store
    }
  }
}

// ---------------- host launcher --------------------------------------------
extern "C" void kernel_launch(void* const* d_in, const int* in_sizes, int n_in,
                              void* d_out, int out_size, void* d_ws, size_t ws_size,
                              hipStream_t stream) {
  const float* x     = (const float*)d_in[0];
  const float* wproj = (const float*)d_in[1];
  const float* bproj = (const float*)d_in[2];
  const float* wout  = (const float*)d_in[3];
  const float* bout  = (const float*)d_in[4];
  // d_in[5] = n_heads (==4), baked in as a compile-time constant
  float* out = (float*)d_out;

  char* ws = (char*)d_ws;
  size_t off = 0;
  auto carve = [&](size_t bytes) -> char* {
    char* p = ws + off;
    off += (bytes + 255) & ~(size_t)255;
    return p;
  };
  _Float16* wph = (_Float16*)carve((size_t)NQKV * CDIM * sizeof(_Float16));
  _Float16* woh = (_Float16*)carve((size_t)CDIM * CDIM * sizeof(_Float16));
  _Float16* qh  = (_Float16*)carve((size_t)BATCH * NH * SEQ * DK * sizeof(_Float16));
  _Float16* kh  = (_Float16*)carve((size_t)BATCH * NH * SEQ * DK * sizeof(_Float16));
  _Float16* vh  = (_Float16*)carve((size_t)BATCH * NH * SEQ * DK * sizeof(_Float16));
  _Float16* oh  = (_Float16*)carve((size_t)BATCH * SEQ * NH * DK * sizeof(_Float16));
  float* colmax = (float*)carve((size_t)BATCH * NH * SEQ * sizeof(float));
  float* colsum = (float*)carve((size_t)BATCH * NH * SEQ * sizeof(float));

  // pass 1: weights -> f16
  cvt_weights<<<dim3((NQKV * CDIM + 255) / 256), dim3(256), 0, stream>>>(
      wproj, wout, wph, woh);

  // pass 2: QKV projection  (M=16384, N=1536, K=512), 16x64 tile per wave
  qkv_proj<<<dim3(BATCH * SEQ / 16, NQKV / 256), dim3(128), 0, stream>>>(
      x, wph, bproj, qh, kh, vh);

  // pass 3: per-(b,h,j) column stats of S
  attn_stats<<<dim3(BATCH * NH, SEQ / 64), dim3(128), 0, stream>>>(
      qh, kh, colmax, colsum);

  // pass 4: O = P @ V  (per (b,h): 1024x128, K=1024), 16x128 strip per wave
  attn_pv<<<dim3(SEQ / 64, BATCH * NH), dim3(128), 0, stream>>>(
      qh, kh, vh, colmax, colsum, oh);

  // pass 5: out projection + residual  (M=16384, N=512, K=512), 16x64 per wave
  out_proj<<<dim3(BATCH * SEQ / 16, CDIM / 256), dim3(128), 0, stream>>>(
      oh, woh, bout, x, out);
}
